// MACHAttention_58531814310177
// MI455X (gfx1250) — matrix operand order repaired and checked by reading, not compiled, not run on hardware
//
#include <hip/hip_runtime.h>
#include <cstdint>
#include <cstddef>

typedef __attribute__((ext_vector_type(16))) __bf16 v16bf;
typedef __attribute__((ext_vector_type(8)))  float  v8f;
typedef __attribute__((ext_vector_type(4)))  unsigned int v4u;
typedef __attribute__((ext_vector_type(8)))  unsigned int v8u;
typedef unsigned int u32;
typedef unsigned long long u64;

#define DKD   1024
#define HEADS 16
#define HDIM  64
#define TSEQ  4096
#define BSZ   4
#define SBUK  128
#define ROWS  (BSZ * TSEQ)   // 16384

union FragBF { v16bf v; uint4 q[2]; };

static __device__ __forceinline__ __bf16 tobf(float f) {
  union { float f; u32 u; } c; c.f = f;
  u32 r = c.u + 0x7fffu + ((c.u >> 16) & 1u);   // round-to-nearest-even
  union { unsigned short s; __bf16 b; } o; o.s = (unsigned short)(r >> 16);
  return o.b;
}

// A fragment (16x32, MxK): lane m=L%16, hi=L/16; elems 0..7 -> K=k0+hi*8..+7,
// elems 8..15 -> K=k0+16+hi*8..+7  (per ISA 16-bit A layout)
static __device__ __forceinline__ v16bf loadA(const __bf16* row, int k0, int hi) {
  FragBF f;
  f.q[0] = *reinterpret_cast<const uint4*>(row + k0 + hi * 8);
  f.q[1] = *reinterpret_cast<const uint4*>(row + k0 + 16 + hi * 8);
  return f.v;
}
// B fragment (fed as N-major rows of B^T): lane n=L%16, hi=L/16; elems 0..7 -> K=k0+hi*8..+7,
// elems 8..15 -> K=k0+16+hi*8..+7 (same as A)
static __device__ __forceinline__ v16bf loadB(const __bf16* row, int k0, int hi) {
  FragBF f;
  f.q[0] = *reinterpret_cast<const uint4*>(row + k0 + hi * 8);
  f.q[1] = *reinterpret_cast<const uint4*>(row + k0 + 16 + hi * 8);
  return f.v;
}

static __device__ __forceinline__ v8f wmma_bf16(v16bf a, v16bf b, v8f c) {
  return __builtin_amdgcn_wmma_f32_16x16x32_bf16(false, a, false, b, (short)0, c, false, false);
}

// TDM: 2D tile (tile_k contiguous elems per row, tile_rows rows, rowStride elems),
// 2-byte elements, into LDS at ldsOff. Descriptor per CDNA5 ISA.
static __device__ __forceinline__ void tdm_load_tile_2d(u32 ldsOff, const void* g,
                                                        u32 tile_k, u32 tile_rows,
                                                        u32 rowStrideElems) {
  v4u g0; v8u g1;
  u64 ga = (u64)(uintptr_t)g;
  g0[0] = 1u;                                           // count=1 valid descriptor
  g0[1] = ldsOff;                                       // lds_addr bytes
  g0[2] = (u32)ga;                                      // global_addr[31:0]
  g0[3] = (u32)((ga >> 32) & 0x1FFFFFFull) | (2u << 30);// addr[56:32] | type=2
  g1[0] = (1u << 16);                                   // data_size = 2 bytes
  g1[1] = (tile_k & 0xffffu) << 16;                     // tensor_dim0 lo16
  g1[2] = (tile_k >> 16) | ((tile_rows & 0xffffu) << 16);// dim0 hi | dim1 lo16
  g1[3] = (tile_rows >> 16) | ((tile_k & 0xffffu) << 16);// dim1 hi | tile_dim0
  g1[4] = tile_rows & 0xffffu;                          // tile_dim1 (tile_dim2=0)
  g1[5] = rowStrideElems;                               // tensor_dim0_stride lo32
  g1[6] = 0u;
  g1[7] = 0u;
  asm volatile("tensor_load_to_lds %0, %1" :: "s"(g0), "s"(g1) : "memory");
}

// ---------------------------------------------------------------- converts
__global__ void __launch_bounds__(256) mach_cvt_bf16(const float* __restrict__ src,
                                                     __bf16* __restrict__ dst, int n) {
  for (int i = blockIdx.x * 256 + threadIdx.x; i < n; i += gridDim.x * 256)
    dst[i] = tobf(src[i]);
}

// ------------------------------------------------------------ GEMM core body
// 128x128 block tile, 8 waves x (16m x 128n), K=1024.
// W tile double-buffered in LDS (128 rows x 64 k, 16KB each) via TDM; two
// WMMA sub-steps per buffer (k-unroll x2 -> no accumulator copy churn).
static __device__ __forceinline__ void gemm_128x128_tdm(
    const __bf16* __restrict__ aRow,   // A row for this lane (pitch DKD)
    const __bf16* __restrict__ Wtile,  // W + n0*DKD (row-major, pitch DKD)
    __bf16* ldsB,                      // [2][128*64]
    int hi, int ml, v8f acc[8]) {
  const int NIT = DKD / 64;            // 16
  if (threadIdx.x < 32)
    tdm_load_tile_2d((u32)(uintptr_t)ldsB, Wtile, 64u, 128u, (u32)DKD);
  for (int it = 0; it < NIT; ++it) {
    const __bf16* cur = ldsB + (size_t)(it & 1) * (128 * 64);
    __syncthreads();   // prior reads of the write buffer are complete
    if (threadIdx.x < 32) {
      if (it + 1 < NIT) {
        tdm_load_tile_2d((u32)(uintptr_t)(ldsB + (size_t)((it + 1) & 1) * (128 * 64)),
                         Wtile + (it + 1) * 64, 64u, 128u, (u32)DKD);
        __builtin_amdgcn_s_wait_tensorcnt(1);   // tile `it` done (TDM is in-order)
      } else {
        __builtin_amdgcn_s_wait_tensorcnt(0);
      }
    }
    __syncthreads();   // publish completed tile to all waves
#pragma unroll
    for (int ks = 0; ks < 64; ks += 32) {
      v16bf a = loadA(aRow, it * 64 + ks, hi);
      v16bf bfr[8];
#pragma unroll
      for (int nt = 0; nt < 8; ++nt)
        bfr[nt] = loadB(cur + (size_t)(nt * 16 + ml) * 64, ks, hi);
#pragma unroll
      for (int nt = 0; nt < 8; ++nt)
        acc[nt] = wmma_bf16(a, bfr[nt], acc[nt]);
    }
  }
}

// ---------------------------------------------------------------- QKV GEMM
__global__ void __launch_bounds__(256) mach_gemm_qkv(
    const __bf16* __restrict__ xh,
    const __bf16* __restrict__ wqh, const __bf16* __restrict__ wkh,
    const __bf16* __restrict__ wvh,
    __bf16* __restrict__ qh, float* __restrict__ kf,
    __bf16* __restrict__ kT, __bf16* __restrict__ vT) {
  __shared__ __align__(16) __bf16 ldsB[2 * 128 * 64];
  const int lane = threadIdx.x & 31, wave = threadIdx.x >> 5;
  const int hi = lane >> 4, ml = lane & 15;
  const int n0 = blockIdx.x * 128;
  const int m0 = blockIdx.y * 128 + wave * 16;
  const int sel = blockIdx.z;
  const __bf16* W = (sel == 0) ? wqh : (sel == 1) ? wkh : wvh;
  v8f acc[8] = {};
  gemm_128x128_tdm(xh + (size_t)(m0 + ml) * DKD, W + (size_t)n0 * DKD,
                   ldsB, hi, ml, acc);
#pragma unroll
  for (int nt = 0; nt < 8; ++nt)
#pragma unroll
    for (int r = 0; r < 8; ++r) {
      int row = m0 + r + 8 * hi;        // (b,t)
      int col = n0 + nt * 16 + ml;      // h*64+d
      float v = acc[nt][r];
      int b = row >> 12, t = row & (TSEQ - 1);
      int h = col >> 6, d = col & 63;
      size_t bhtd = ((size_t)(b * HEADS + h) * TSEQ + t) * HDIM + d;
      size_t bhdt = ((size_t)(b * HEADS + h) * HDIM + d) * TSEQ + t;
      if (sel == 0) qh[bhtd] = tobf(v);
      else if (sel == 1) { kf[bhtd] = v; kT[bhdt] = tobf(v); }
      else vT[bhdt] = tobf(v);
    }
}

// ---------------------------------------------------------------- output GEMM
__global__ void __launch_bounds__(256) mach_gemm_out(
    const __bf16* __restrict__ ah, const __bf16* __restrict__ woh,
    const float* __restrict__ bo, float* __restrict__ out) {
  __shared__ __align__(16) __bf16 ldsB[2 * 128 * 64];
  const int lane = threadIdx.x & 31, wave = threadIdx.x >> 5;
  const int hi = lane >> 4, ml = lane & 15;
  const int n0 = blockIdx.x * 128;
  const int m0 = blockIdx.y * 128 + wave * 16;
  v8f acc[8] = {};
  gemm_128x128_tdm(ah + (size_t)(m0 + ml) * DKD, woh + (size_t)n0 * DKD,
                   ldsB, hi, ml, acc);
#pragma unroll
  for (int nt = 0; nt < 8; ++nt)
#pragma unroll
    for (int r = 0; r < 8; ++r) {
      int row = m0 + r + 8 * hi;
      int col = n0 + nt * 16 + ml;
      out[(size_t)row * DKD + col] = acc[nt][r] + bo[col];
    }
}

// ---------------------------------------------------------------- soft-hash probs
__global__ void __launch_bounds__(256) mach_probs(
    const float* __restrict__ kf, const float* __restrict__ planesT,
    const float* __restrict__ protosT, const float* __restrict__ lt,
    __bf16* __restrict__ probsT) {
  int idx = blockIdx.x * 256 + threadIdx.x;       // bh*4096 + t
  int t = idx & (TSEQ - 1);
  int bh = idx >> 12;
  const float* kr = kf + (size_t)idx * HDIM;
  float scale = __expf(lt[0]);
  scale = fminf(fmaxf(scale, 0.01f), 20.0f);
  float krow[HDIM];
#pragma unroll
  for (int d = 0; d < HDIM; ++d) krow[d] = kr[d];
  float th[32];
  for (int p = 0; p < 32; ++p) {
    float s = 0.f;
#pragma unroll
    for (int d = 0; d < HDIM; ++d) s += krow[d] * planesT[d * 32 + p];
    th[p] = tanhf(s) / scale;
  }
  __bf16* outp = probsT + (size_t)bh * SBUK * TSEQ + t;
  for (int l = 0; l < 8; ++l) {
    float lg[16]; float mx = -3.0e38f;
    for (int r = 0; r < 16; ++r) {
      float s = th[l * 4 + 0] * protosT[r]       + th[l * 4 + 1] * protosT[16 + r]
              + th[l * 4 + 2] * protosT[32 + r]  + th[l * 4 + 3] * protosT[48 + r];
      lg[r] = s; mx = fmaxf(mx, s);
    }
    float sum = 0.f;
    for (int r = 0; r < 16; ++r) { lg[r] = __expf(lg[r] - mx); sum += lg[r]; }
    float inv = 1.0f / sum;
    for (int r = 0; r < 16; ++r)
      outp[(size_t)(l * 16 + r) * TSEQ] = tobf(lg[r] * inv);
  }
}

// ---------------------------------------------------------------- bucket GEMMs
// bucket[s][d] = sum_t probsT[s][t] * {K,V}^T[d][t]; M=128, N=64, K=4096.
__global__ void __launch_bounds__(256) mach_bucket(
    const __bf16* __restrict__ probsT, const __bf16* __restrict__ kT,
    const __bf16* __restrict__ vT,
    __bf16* __restrict__ bK, __bf16* __restrict__ bVT) {
  const int lane = threadIdx.x & 31, wave = threadIdx.x >> 5;
  const int hi = lane >> 4, ml = lane & 15;
  const int bh = blockIdx.y, sel = blockIdx.x;
  const __bf16* aRow = probsT + (size_t)bh * SBUK * TSEQ + (size_t)(wave * 16 + ml) * TSEQ;
  const __bf16* Bm = ((sel == 0) ? kT : vT) + (size_t)bh * HDIM * TSEQ;
  v8f acc[4] = {};
  for (int t0 = 0; t0 < TSEQ; t0 += 64) {   // k-unroll x2
    __builtin_prefetch(aRow + t0 + 256, 0, 0);
#pragma unroll
    for (int ks = 0; ks < 64; ks += 32) {
      v16bf a = loadA(aRow, t0 + ks, hi);
      v16bf bfr[4];
#pragma unroll
      for (int nt = 0; nt < 4; ++nt)
        bfr[nt] = loadB(Bm + (size_t)(nt * 16 + ml) * TSEQ, t0 + ks, hi);
#pragma unroll
      for (int nt = 0; nt < 4; ++nt)
        acc[nt] = wmma_bf16(a, bfr[nt], acc[nt]);
    }
  }
#pragma unroll
  for (int nt = 0; nt < 4; ++nt)
#pragma unroll
    for (int r = 0; r < 8; ++r) {
      int s = wave * 16 + r + 8 * hi;
      int d = nt * 16 + ml;
      float v = acc[nt][r];
      if (sel == 0) bK[(size_t)bh * SBUK * HDIM + (size_t)s * HDIM + d] = tobf(v);
      else          bVT[(size_t)bh * HDIM * SBUK + (size_t)d * SBUK + s] = tobf(v);
    }
}

// ---------------------------------------------------------------- attention
// scores = Q @ bK^T / 8 -> row softmax (S=128) -> out = P @ bV ; 128 t-rows/block.
__global__ void __launch_bounds__(256) mach_attn(
    const __bf16* __restrict__ qh, const __bf16* __restrict__ bK,
    const __bf16* __restrict__ bVT, __bf16* __restrict__ aout) {
  __shared__ __align__(16) __bf16 P[128 * 128];
  __shared__ float rsum[128];
  const int lane = threadIdx.x & 31, wave = threadIdx.x >> 5;
  const int hi = lane >> 4, ml = lane & 15;
  const int bh = blockIdx.y;
  const int t0 = blockIdx.x * 128;
  const int b = bh >> 4, h = bh & 15;
  const __bf16* Q  = qh + ((size_t)bh * TSEQ + (t0 + wave * 16 + ml)) * HDIM;
  const __bf16* BK = bK + (size_t)bh * SBUK * HDIM;
  const __bf16* BV = bVT + (size_t)bh * HDIM * SBUK;

  v8f sc[8] = {};
#pragma unroll
  for (int k0 = 0; k0 < HDIM; k0 += 32) {
    v16bf a = loadA(Q, k0, hi);
    v16bf bfr[8];
#pragma unroll
    for (int nt = 0; nt < 8; ++nt)
      bfr[nt] = loadB(BK + (size_t)(nt * 16 + ml) * HDIM, k0, hi);
#pragma unroll
    for (int nt = 0; nt < 8; ++nt)
      sc[nt] = wmma_bf16(a, bfr[nt], sc[nt]);
  }
  // row softmax: row m=r+8*hi lives across the 16 lanes of a half-wave + 8 n-tiles
#pragma unroll
  for (int r = 0; r < 8; ++r) {
    float mx = -3.0e38f;
#pragma unroll
    for (int nt = 0; nt < 8; ++nt) { sc[nt][r] *= 0.125f; mx = fmaxf(mx, sc[nt][r]); }
#pragma unroll
    for (int off = 1; off < 16; off <<= 1) mx = fmaxf(mx, __shfl_xor(mx, off, 32));
    float sum = 0.f;
#pragma unroll
    for (int nt = 0; nt < 8; ++nt) { float e = __expf(sc[nt][r] - mx); sc[nt][r] = e; sum += e; }
#pragma unroll
    for (int off = 1; off < 16; off <<= 1) sum += __shfl_xor(sum, off, 32);
    int m = wave * 16 + r + 8 * hi;
#pragma unroll
    for (int nt = 0; nt < 8; ++nt) P[(size_t)m * 128 + nt * 16 + ml] = tobf(sc[nt][r]);
    if (ml == 0) rsum[m] = sum;
  }
  __syncthreads();

  v8f o[4] = {};
  const __bf16* aRow = P + (size_t)(wave * 16 + ml) * 128;
#pragma unroll
  for (int s0 = 0; s0 < SBUK; s0 += 32) {
    v16bf a = loadA(aRow, s0, hi);
    v16bf bfr[4];
#pragma unroll
    for (int nt = 0; nt < 4; ++nt)
      bfr[nt] = loadB(BV + (size_t)(nt * 16 + ml) * SBUK, s0, hi);
#pragma unroll
    for (int nt = 0; nt < 4; ++nt)
      o[nt] = wmma_bf16(a, bfr[nt], o[nt]);
  }
#pragma unroll
  for (int nt = 0; nt < 4; ++nt)
#pragma unroll
    for (int r = 0; r < 8; ++r) {
      int m = wave * 16 + r + 8 * hi;
      int trow = t0 + m;
      int d = nt * 16 + ml;
      float v = o[nt][r] / rsum[m];
      aout[((size_t)b * TSEQ + trow) * DKD + h * HDIM + d] = tobf(v);
    }
}

// ---------------------------------------------------------------- launch
extern "C" void kernel_launch(void* const* d_in, const int* in_sizes, int n_in,
                              void* d_out, int out_size, void* d_ws, size_t ws_size,
                              hipStream_t stream) {
  const float* x       = (const float*)d_in[0];
  const float* Wq      = (const float*)d_in[1];
  const float* Wk      = (const float*)d_in[2];
  const float* Wv      = (const float*)d_in[3];
  const float* Wo      = (const float*)d_in[4];
  const float* bo      = (const float*)d_in[5];
  const float* planesT = (const float*)d_in[6];
  const float* protosT = (const float*)d_in[7];
  const float* lt      = (const float*)d_in[8];

  char* ws = (char*)d_ws;
  size_t off = 0;
  auto take = [&](size_t bytes) {
    char* p = ws + off; off += (bytes + 255) & ~(size_t)255; return p;
  };
  __bf16* xh   = (__bf16*)take((size_t)ROWS * DKD * 2);        // 32 MB
  __bf16* wqh  = (__bf16*)take((size_t)DKD * DKD * 2);         // 2 MB
  __bf16* wkh  = (__bf16*)take((size_t)DKD * DKD * 2);
  __bf16* wvh  = (__bf16*)take((size_t)DKD * DKD * 2);
  __bf16* woh  = (__bf16*)take((size_t)DKD * DKD * 2);
  __bf16* qh   = (__bf16*)take((size_t)ROWS * DKD * 2);        // 32 MB (b,h,t,d)
  float*  kf   = (float*) take((size_t)ROWS * DKD * 4);        // 64 MB (b,h,t,d)
  __bf16* kT   = (__bf16*)take((size_t)ROWS * DKD * 2);        // 32 MB (b,h,d,t)
  __bf16* vT   = (__bf16*)take((size_t)ROWS * DKD * 2);        // 32 MB (b,h,d,t)
  __bf16* prT  = (__bf16*)take((size_t)BSZ * HEADS * SBUK * TSEQ * 2); // 64 MB
  __bf16* bK   = (__bf16*)take((size_t)BSZ * HEADS * SBUK * HDIM * 2); // 1 MB
  __bf16* bVT  = (__bf16*)take((size_t)BSZ * HEADS * HDIM * SBUK * 2); // 1 MB
  __bf16* aout = (__bf16*)take((size_t)ROWS * DKD * 2);        // 32 MB

  mach_cvt_bf16<<<4096, 256, 0, stream>>>(x,  xh,  ROWS * DKD);
  mach_cvt_bf16<<<1024, 256, 0, stream>>>(Wq, wqh, DKD * DKD);
  mach_cvt_bf16<<<1024, 256, 0, stream>>>(Wk, wkh, DKD * DKD);
  mach_cvt_bf16<<<1024, 256, 0, stream>>>(Wv, wvh, DKD * DKD);
  mach_cvt_bf16<<<1024, 256, 0, stream>>>(Wo, woh, DKD * DKD);

  mach_gemm_qkv<<<dim3(DKD / 128, ROWS / 128, 3), 256, 0, stream>>>(
      xh, wqh, wkh, wvh, qh, kf, kT, vT);

  mach_probs<<<(BSZ * HEADS * TSEQ) / 256, 256, 0, stream>>>(
      kf, planesT, protosT, lt, prT);

  mach_bucket<<<dim3(2, BSZ * HEADS), 256, 0, stream>>>(prT, kT, vT, bK, bVT);

  mach_attn<<<dim3(TSEQ / 128, BSZ * HEADS), 256, 0, stream>>>(qh, bK, bVT, aout);

  mach_gemm_out<<<dim3(DKD / 128, ROWS / 128), 256, 0, stream>>>(
      aout, woh, bo, (float*)d_out);
}
